// LocalLoraMLP_3796751090125
// MI455X (gfx1250) — matrix operand order repaired and checked by reading, not compiled, not run on hardware
//
#include <hip/hip_runtime.h>

typedef __attribute__((ext_vector_type(16))) _Float16 v16h;
typedef __attribute__((ext_vector_type(8)))  _Float16 v8h;
typedef __attribute__((ext_vector_type(4)))  _Float16 v4h;
typedef __attribute__((ext_vector_type(8)))  float    v8f;
typedef __attribute__((ext_vector_type(4)))  float    v4f;

#define S_ 2048
#define H_ 2048
#define I_ 5632
#define M_ 3

// ---------- helpers ----------

static __device__ __forceinline__ v16h combine8(v8h lo, v8h hi) {
  v16h r;
#pragma unroll
  for (int i = 0; i < 8; ++i) { r[i] = lo[i]; r[i + 8] = hi[i]; }
  return r;
}

static __device__ __forceinline__ v8f zero_v8f() {
  v8f z;
#pragma unroll
  for (int i = 0; i < 8; ++i) z[i] = 0.0f;
  return z;
}

static __device__ __forceinline__ v8h zero_v8h() {
  v8h z;
#pragma unroll
  for (int i = 0; i < 8; ++i) z[i] = (_Float16)0.0f;
  return z;
}

// A-matrix fragment, 16x32 f16 (M x K), per-lane contiguous K:
// lane l (l<16): M=l, K = 0..15
// lane l (l>=16): M=l-16, K = 16..31
static __device__ __forceinline__ v16h frag_a(const _Float16* __restrict__ A,
                                              size_t ld, int row0, int k0) {
  int l = threadIdx.x & 31;
  int m = l & 15;
  int kb = (l < 16) ? 0 : 16;
  const _Float16* p = A + (size_t)(row0 + m) * ld + (size_t)(k0 + kb);
  v8h lo = *(const v8h*)p;          // K = kb .. kb+7
  v8h hi = *(const v8h*)(p + 8);    // K = kb+8 .. kb+15
  return combine8(lo, hi);
}

// B-matrix fragment, 32x16 f16 (K x N), column n = lane&15, per-lane contiguous K.
// lanes 0-15: K=0..15 of column n; lanes 16-31: K=16..31 of column n.
// B stored as [N, K] row-major (weights are [out, in] with in contiguous).
static __device__ __forceinline__ v16h frag_b(const _Float16* __restrict__ B,
                                              size_t ld, int col0, int k0) {
  int l = threadIdx.x & 31;
  int n = l & 15;
  int kb = (l < 16) ? 0 : 16;
  const _Float16* p = B + (size_t)(col0 + n) * ld + (size_t)(k0 + kb);
  v8h lo = *(const v8h*)p;
  v8h hi = *(const v8h*)(p + 8);
  return combine8(lo, hi);
}

// B fragment for rank-16 LoRA update: B stored [N,16]; K=16..31 are zero.
static __device__ __forceinline__ v16h frag_b16(const _Float16* __restrict__ B,
                                                int col0) {
  int l = threadIdx.x & 31;
  int n = l & 15;
  v8h lo = zero_v8h();
  v8h hi = zero_v8h();
  if (l < 16) {
    const _Float16* p = B + (size_t)(col0 + n) * 16;
    lo = *(const v8h*)p;
    hi = *(const v8h*)(p + 8);
  }
  return combine8(lo, hi);
}

static __device__ __forceinline__ v8f wmma_f16(v16h a, v16h b, v8f c) {
  return __builtin_amdgcn_wmma_f32_16x16x32_f16(
      /*neg_a=*/false, a, /*neg_b=*/false, b,
      /*c_mod=*/(short)0, c, /*reuse_a=*/false, /*reuse_b=*/false);
}

// ---------- kernels ----------

// fp32 -> f16 with scale, 4 elements/thread iteration (all sizes are %4 == 0)
__global__ void cvt4_kernel(const float* __restrict__ s,
                            _Float16* __restrict__ d, size_t n4, float scale) {
  size_t i = (size_t)blockIdx.x * blockDim.x + threadIdx.x;
  size_t st = (size_t)gridDim.x * blockDim.x;
  const v4f* S = (const v4f*)s;
  v4h* D = (v4h*)d;
  for (; i < n4; i += st) {
    v4f v = S[i];
    v4h o;
#pragma unroll
    for (int j = 0; j < 4; ++j) o[j] = (_Float16)(v[j] * scale);
    D[i] = o;
  }
}

// t[m][s, 0:16] = X_m[s,:] @ P_m[r,:]^T  (P stored [M,16,K]); cols 16..31 zeroed.
__global__ void rank16_kernel(const _Float16* __restrict__ X, size_t ld,
                              size_t x_mstride,
                              const _Float16* __restrict__ P,
                              _Float16* __restrict__ T, int K) {
  int wid = blockIdx.x * (blockDim.x >> 5) + (threadIdx.x >> 5);
  if (wid >= M_ * (S_ / 16)) return;
  int m = wid / (S_ / 16);
  int s0 = (wid % (S_ / 16)) * 16;
  const _Float16* Xm = X + (size_t)m * x_mstride;
  const _Float16* Pm = P + (size_t)m * 16 * ld;
  _Float16* Tm = T + (size_t)m * S_ * 32;

  v8f acc = zero_v8f();
#pragma unroll 4
  for (int k = 0; k < K; k += 32) {
    v16h va = frag_a(Xm, ld, s0, k);
    v16h vb = frag_b(Pm, ld, 0, k);
    acc = wmma_f16(va, vb, acc);
  }
  int l = threadIdx.x & 31;
  int n = l & 15;
  int rb = (l < 16) ? 0 : 8;
#pragma unroll
  for (int j = 0; j < 8; ++j) {
    size_t off = (size_t)(s0 + j + rb) * 32;
    Tm[off + n] = (_Float16)acc[j];
    Tm[off + n + 16] = (_Float16)0.0f;
  }
}

// h[m][s,i] = silu(x@Wg^T + tg_m@(2*Bg_m)^T) * (x@Wu^T + tu_m@(2*Bu_m)^T)
// Each wave: one (m, 32-row, 64-col) macro-tile; gate & up share both A
// fragments, every B fragment is reused across the two row sub-tiles.
__global__ void gateup_kernel(const _Float16* __restrict__ x16,
                              const _Float16* __restrict__ Wg16,
                              const _Float16* __restrict__ Wu16,
                              const _Float16* __restrict__ Bg16,  // [M,I,16], pre-scaled
                              const _Float16* __restrict__ Bu16,  // [M,I,16], pre-scaled
                              const _Float16* __restrict__ tg,    // [M,S,32]
                              const _Float16* __restrict__ tu,    // [M,S,32]
                              _Float16* __restrict__ h16) {       // [M,S,I]
  const int TOTAL = M_ * (S_ / 32) * (I_ / 64);
  int wid = blockIdx.x * (blockDim.x >> 5) + (threadIdx.x >> 5);
  if (wid >= TOTAL) return;
  int m = wid / ((S_ / 32) * (I_ / 64));
  int rem = wid % ((S_ / 32) * (I_ / 64));
  int s0 = (rem / (I_ / 64)) * 32;
  int n0 = (rem % (I_ / 64)) * 64;

  v8f accg[2][4], accu[2][4];
#pragma unroll
  for (int r = 0; r < 2; ++r)
#pragma unroll
    for (int t = 0; t < 4; ++t) { accg[r][t] = zero_v8f(); accu[r][t] = zero_v8f(); }

#pragma unroll 2
  for (int k = 0; k < H_; k += 32) {
    v16h va0 = frag_a(x16, H_, s0, k);
    v16h va1 = frag_a(x16, H_, s0 + 16, k);
#pragma unroll
    for (int t = 0; t < 4; ++t) {
      v16h vb = frag_b(Wg16, H_, n0 + 16 * t, k);
      accg[0][t] = wmma_f16(va0, vb, accg[0][t]);
      accg[1][t] = wmma_f16(va1, vb, accg[1][t]);
    }
#pragma unroll
    for (int t = 0; t < 4; ++t) {
      v16h vb = frag_b(Wu16, H_, n0 + 16 * t, k);
      accu[0][t] = wmma_f16(va0, vb, accu[0][t]);
      accu[1][t] = wmma_f16(va1, vb, accu[1][t]);
    }
  }

  // LoRA rank-16 epilogue (one extra WMMA per 16x16 tile per branch)
  v16h vag[2], vau[2];
#pragma unroll
  for (int r = 0; r < 2; ++r) {
    vag[r] = frag_a(tg + (size_t)m * S_ * 32, 32, s0 + 16 * r, 0);
    vau[r] = frag_a(tu + (size_t)m * S_ * 32, 32, s0 + 16 * r, 0);
  }
#pragma unroll
  for (int t = 0; t < 4; ++t) {
    v16h vbg = frag_b16(Bg16 + (size_t)m * I_ * 16, n0 + 16 * t);
    v16h vbu = frag_b16(Bu16 + (size_t)m * I_ * 16, n0 + 16 * t);
#pragma unroll
    for (int r = 0; r < 2; ++r) {
      accg[r][t] = wmma_f16(vag[r], vbg, accg[r][t]);
      accu[r][t] = wmma_f16(vau[r], vbu, accu[r][t]);
    }
  }

  int l = threadIdx.x & 31;
  int n = l & 15;
  int rb = (l < 16) ? 0 : 8;
  _Float16* hm = h16 + (size_t)m * S_ * I_;
#pragma unroll
  for (int r = 0; r < 2; ++r) {
#pragma unroll
    for (int t = 0; t < 4; ++t) {
#pragma unroll
      for (int j = 0; j < 8; ++j) {
        float g = accg[r][t][j];
        float u = accu[r][t][j];
        float hv = (g / (1.0f + __expf(-g))) * u;  // silu(g) * u
        hm[(size_t)(s0 + 16 * r + j + rb) * I_ + (size_t)(n0 + 16 * t + n)] =
            (_Float16)hv;
      }
    }
  }
}

// out[s,h'] = sum_m mask[m][s] * ( h_m @ Wd^T + t2_m @ (2*Bd_m)^T )[s,h']
// 32x64 macro-tile per wave; B fragments reused across the two row sub-tiles.
__global__ void down_kernel(const _Float16* __restrict__ h16,   // [M,S,I]
                            const _Float16* __restrict__ Wd16,  // [H,I]
                            const _Float16* __restrict__ Bd16,  // [M,H,16], pre-scaled
                            const _Float16* __restrict__ t2,    // [M,S,32]
                            const float* __restrict__ mask,     // [M,S]
                            float* __restrict__ out) {          // [S,H]
  const int TOTAL = (S_ / 32) * (H_ / 64);
  int wid = blockIdx.x * (blockDim.x >> 5) + (threadIdx.x >> 5);
  if (wid >= TOTAL) return;
  int s0 = (wid / (H_ / 64)) * 32;
  int h0 = (wid % (H_ / 64)) * 64;

  int l = threadIdx.x & 31;
  int n = l & 15;
  int rb = (l < 16) ? 0 : 8;

  v8f acco[2][4];
#pragma unroll
  for (int r = 0; r < 2; ++r)
#pragma unroll
    for (int t = 0; t < 4; ++t) acco[r][t] = zero_v8f();

#pragma unroll 1
  for (int m = 0; m < M_; ++m) {
    const _Float16* hm = h16 + (size_t)m * S_ * I_;
    v8f acc[2][4];
#pragma unroll
    for (int r = 0; r < 2; ++r)
#pragma unroll
      for (int t = 0; t < 4; ++t) acc[r][t] = zero_v8f();

#pragma unroll 2
    for (int k = 0; k < I_; k += 32) {
      v16h va0 = frag_a(hm, I_, s0, k);
      v16h va1 = frag_a(hm, I_, s0 + 16, k);
#pragma unroll
      for (int t = 0; t < 4; ++t) {
        v16h vb = frag_b(Wd16, I_, h0 + 16 * t, k);
        acc[0][t] = wmma_f16(va0, vb, acc[0][t]);
        acc[1][t] = wmma_f16(va1, vb, acc[1][t]);
      }
    }

    // down-proj LoRA epilogue
    v16h va2[2];
#pragma unroll
    for (int r = 0; r < 2; ++r)
      va2[r] = frag_a(t2 + (size_t)m * S_ * 32, 32, s0 + 16 * r, 0);
#pragma unroll
    for (int t = 0; t < 4; ++t) {
      v16h vb2 = frag_b16(Bd16 + (size_t)m * H_ * 16, h0 + 16 * t);
#pragma unroll
      for (int r = 0; r < 2; ++r) acc[r][t] = wmma_f16(va2[r], vb2, acc[r][t]);
    }

    // weighted combine over modalities
#pragma unroll
    for (int r = 0; r < 2; ++r) {
#pragma unroll
      for (int j = 0; j < 8; ++j) {
        float mk = mask[(size_t)m * S_ + (size_t)(s0 + 16 * r + j + rb)];
#pragma unroll
        for (int t = 0; t < 4; ++t) acco[r][t][j] += acc[r][t][j] * mk;
      }
    }
  }

#pragma unroll
  for (int r = 0; r < 2; ++r) {
#pragma unroll
    for (int t = 0; t < 4; ++t) {
#pragma unroll
      for (int j = 0; j < 8; ++j) {
        out[(size_t)(s0 + 16 * r + j + rb) * H_ + (size_t)(h0 + 16 * t + n)] =
            acco[r][t][j];
      }
    }
  }
}

// ---------- host ----------

extern "C" void kernel_launch(void* const* d_in, const int* in_sizes, int n_in,
                              void* d_out, int out_size, void* d_ws, size_t ws_size,
                              hipStream_t stream) {
  (void)in_sizes; (void)n_in; (void)out_size; (void)ws_size;
  const float* x    = (const float*)d_in[0];
  const float* Wg   = (const float*)d_in[1];
  const float* Wu   = (const float*)d_in[2];
  const float* Wd   = (const float*)d_in[3];
  const float* Ag   = (const float*)d_in[4];
  const float* Bg   = (const float*)d_in[5];
  const float* Au   = (const float*)d_in[6];
  const float* Bu   = (const float*)d_in[7];
  const float* Ad   = (const float*)d_in[8];
  const float* Bd   = (const float*)d_in[9];
  const float* mask = (const float*)d_in[10];
  float* out = (float*)d_out;

  const float SCALE = 32.0f / 16.0f;  // 2.0, exact in fp

  char* w = (char*)d_ws;
  auto carve = [&](size_t bytes) -> char* {
    char* p = w;
    w += (bytes + 255) & ~(size_t)255;
    return p;
  };
  _Float16* x16  = (_Float16*)carve((size_t)S_ * H_ * 2);
  _Float16* Wg16 = (_Float16*)carve((size_t)I_ * H_ * 2);
  _Float16* Wu16 = (_Float16*)carve((size_t)I_ * H_ * 2);
  _Float16* Wd16 = (_Float16*)carve((size_t)H_ * I_ * 2);
  _Float16* Ag16 = (_Float16*)carve((size_t)M_ * 16 * H_ * 2);
  _Float16* Au16 = (_Float16*)carve((size_t)M_ * 16 * H_ * 2);
  _Float16* Bg16 = (_Float16*)carve((size_t)M_ * I_ * 16 * 2);
  _Float16* Bu16 = (_Float16*)carve((size_t)M_ * I_ * 16 * 2);
  _Float16* Ad16 = (_Float16*)carve((size_t)M_ * 16 * I_ * 2);
  _Float16* Bd16 = (_Float16*)carve((size_t)M_ * H_ * 16 * 2);
  _Float16* tg   = (_Float16*)carve((size_t)M_ * S_ * 32 * 2);
  _Float16* tu   = (_Float16*)carve((size_t)M_ * S_ * 32 * 2);
  _Float16* t2   = (_Float16*)carve((size_t)M_ * S_ * 32 * 2);
  _Float16* h16  = (_Float16*)carve((size_t)M_ * S_ * I_ * 2);

  auto cvt = [&](const float* s, _Float16* d, size_t n, float sc) {
    size_t n4 = n >> 2;  // all element counts are multiples of 4
    int blocks = (int)((n4 + 1023) / 1024);
    if (blocks > 4096) blocks = 4096;
    cvt4_kernel<<<blocks, 256, 0, stream>>>(s, d, n4, sc);
  };
  cvt(x,  x16,  (size_t)S_ * H_, 1.0f);
  cvt(Wg, Wg16, (size_t)I_ * H_, 1.0f);
  cvt(Wu, Wu16, (size_t)I_ * H_, 1.0f);
  cvt(Wd, Wd16, (size_t)H_ * I_, 1.0f);
  cvt(Ag, Ag16, (size_t)M_ * 16 * H_, 1.0f);
  cvt(Au, Au16, (size_t)M_ * 16 * H_, 1.0f);
  cvt(Bg, Bg16, (size_t)M_ * I_ * 16, SCALE);
  cvt(Bu, Bu16, (size_t)M_ * I_ * 16, SCALE);
  cvt(Ad, Ad16, (size_t)M_ * 16 * I_, 1.0f);
  cvt(Bd, Bd16, (size_t)M_ * H_ * 16, SCALE);

  // t projections: 3*128 = 384 waves (8 waves / 256-thread block)
  rank16_kernel<<<48, 256, 0, stream>>>(x16, (size_t)H_, (size_t)0, Ag16, tg, H_);
  rank16_kernel<<<48, 256, 0, stream>>>(x16, (size_t)H_, (size_t)0, Au16, tu, H_);

  // gate/up + silu fused: 3*64*88 = 16896 waves -> 2112 blocks
  gateup_kernel<<<2112, 256, 0, stream>>>(x16, Wg16, Wu16, Bg16, Bu16, tg, tu, h16);

  // t2 = h_m @ Ad_m^T
  rank16_kernel<<<48, 256, 0, stream>>>(h16, (size_t)I_, (size_t)S_ * I_, Ad16, t2, I_);

  // down + LoRA + mask-combine: 64*32 = 2048 waves -> 256 blocks
  down_kernel<<<256, 256, 0, stream>>>(h16, Wd16, Bd16, t2, mask, out);
}
